// GPT_31507880083557
// MI455X (gfx1250) — matrix operand order repaired and checked by reading, not compiled, not run on hardware
//
#include <hip/hip_runtime.h>

// ---------------------------------------------------------------------------
// Problem constants (fixed by reference: B=4, L=8192, D=1024, C=L/4=2048)
// ---------------------------------------------------------------------------
constexpr int B_ = 4;
constexpr int L_ = 8192;
constexpr int D_ = 1024;
constexpr int C_ = 2048;

constexpr int MTILE = 32;         // tokens per workgroup (2 M-subtiles/wave)
constexpr int NROWS = MTILE + 1;  // staged x rows (need l0-1 .. l0+MTILE-1)

typedef __attribute__((ext_vector_type(16))) __bf16 v16bf;
typedef __attribute__((ext_vector_type(8)))  __bf16 v8bf;
typedef __attribute__((ext_vector_type(8)))  float  v8f;

#define WMMA_BF16(a, b, c)                                                     \
  __builtin_amdgcn_wmma_f32_16x16x32_bf16(false, (a), false, (b), (short)0,    \
                                          (c), false, false)

// A-fragment (16-bit A 16x32, ISA 7.12.2): lane holds 8 contiguous K at +0 and
// 8 contiguous K at +16 (K-base 0 for lanes 0-15, 8 for lanes 16-31).
__device__ __forceinline__ v16bf ld_a(const __bf16* p) {
  v8bf lo = *(const v8bf*)(p);
  v8bf hi = *(const v8bf*)(p + 16);
  v16bf r;
#pragma unroll
  for (int i = 0; i < 8; ++i) { r[i] = lo[i]; r[i + 8] = hi[i]; }
  return r;
}
// B-fragment (16-bit B 32x16): lane n%16 holds K = 8*(lane/16) + {0..7} and
// 16 + 8*(lane/16) + {0..7} of one W row (W is row-major [e][d]).
__device__ __forceinline__ v16bf ld_b(const __bf16* p) {
  v8bf lo = *(const v8bf*)(p);
  v8bf hi = *(const v8bf*)(p + 16);
  v16bf r;
#pragma unroll
  for (int i = 0; i < 8; ++i) { r[i] = lo[i]; r[i + 8] = hi[i]; }
  return r;
}

// ---------------------------------------------------------------------------
// Kernel 1: split Wq/Wk (fp32) into bf16 hi/lo pairs
// ---------------------------------------------------------------------------
__global__ void k_wsplit(const float* __restrict__ Wq, const float* __restrict__ Wk,
                         __bf16* __restrict__ Wqh, __bf16* __restrict__ Wql,
                         __bf16* __restrict__ Wkh, __bf16* __restrict__ Wkl) {
  int i = blockIdx.x * 256 + threadIdx.x;
  if (i >= D_ * D_) return;
  float v = Wq[i];
  __bf16 h = (__bf16)v;
  Wqh[i] = h;
  Wql[i] = (__bf16)(v - (float)h);
  v = Wk[i];
  h = (__bf16)v;
  Wkh[i] = h;
  Wkl[i] = (__bf16)(v - (float)h);
}

// ---------------------------------------------------------------------------
// Kernel 2: fused WMMA GEMM -> boundary probability p[b,l]
//   p[b,0]=1 ; p[b,l] = clip(0.5*(1 - cos(Wq x[l], Wk x[l-1])), 0, 1)
// One WG = 16 waves covers MTILE=32 tokens (2 M-subtiles per wave); each wave
// owns 4 N-tiles (64 e-cols). bf16 hi/lo 3-product GEMM; each B fragment is
// reused across both M-subtiles to halve L2 traffic for W.
//
// LDS layout: hi/lo interleaved per row -> sX[row][0=hi|1=lo][LDSROW]. The 8
// A-fragments of each M-subtile span < 6.3 KB from that subtile's base
// pointer, so every ds_load folds into a 16-bit immediate offset. An inline
// asm value-barrier keeps the two subtile bases as independent running VGPR
// pointers (otherwise LLVM rewrites base1 = base0 + 66048, which exceeds the
// DS immediate field and forces per-load VALU adds + WMMA hazard NOPs).
// ---------------------------------------------------------------------------
constexpr int LDSROW = D_ + 8;  // 1032 elems; hi/lo pair stride 4128 B (16B-aligned)

__global__ __launch_bounds__(512) void k_gemm_p(
    const float* __restrict__ x,
    const __bf16* __restrict__ Wqh, const __bf16* __restrict__ Wql,
    const __bf16* __restrict__ Wkh, const __bf16* __restrict__ Wkl,
    float* __restrict__ p_full) {
  __shared__ __bf16 sX[NROWS][2][LDSROW];
  __shared__ float sSq[MTILE], sSk[MTILE], sSqk[MTILE];

  const int wg  = blockIdx.x;            // 0 .. B*(L/MTILE)-1
  const int b   = wg / (L_ / MTILE);
  const int l0  = (wg % (L_ / MTILE)) * MTILE;
  const int tid = threadIdx.x;

  // Stage x rows [l0-1 .. l0+MTILE-1] into LDS, split fp32 -> bf16 hi + lo.
  for (int i = tid; i < NROWS * D_; i += 512) {
    int r = i >> 10, c = i & (D_ - 1);
    int l = l0 - 1 + r;
    if (l < 0) l = 0;  // token 0's p is forced to 1 below; value irrelevant
    float v = x[((size_t)b * L_ + l) * D_ + c];
    __bf16 h = (__bf16)v;
    sX[r][0][c] = h;
    sX[r][1][c] = (__bf16)(v - (float)h);
  }
  if (tid < MTILE) { sSq[tid] = 0.f; sSk[tid] = 0.f; sSqk[tid] = 0.f; }
  __syncthreads();

  const int wave = tid >> 5;
  const int lane = tid & 31;
  const int lm   = lane & 15;   // A row (token) / B column (e)
  const int lhi  = lane >> 4;   // half-wave

  float psq[2][8], psk[2][8], psqk[2][8];
#pragma unroll
  for (int s = 0; s < 2; ++s)
#pragma unroll
    for (int j = 0; j < 8; ++j) { psq[s][j] = 0.f; psk[s][j] = 0.f; psqk[s][j] = 0.f; }

  // Constant element offsets from a per-subtile base pointer (&sX[row0][0][aoff]):
  //   +0          : k-matrix hi  (row row0,   x[l-1])
  //   +LDSROW     : k-matrix lo
  //   +2*LDSROW   : q-matrix hi  (row row0+1, x[l])
  //   +3*LDSROW   : q-matrix lo
  const int aoff0 = lhi * 8;
  int row1 = lm + 16;
  asm volatile("" : "+v"(row1));  // opaque: keep subtile-1 base independent

  for (int nt = 0; nt < 4; ++nt) {            // 4 N-tiles (64 e-cols) per wave
    const int e0 = (wave * 4 + nt) * 16;
    const size_t wrow = (size_t)(e0 + lm) * D_ + (size_t)lhi * 8;
    const __bf16* pqh = Wqh + wrow;
    const __bf16* pql = Wql + wrow;
    const __bf16* pkh = Wkh + wrow;
    const __bf16* pkl = Wkl + wrow;

    const __bf16* aB0 = &sX[lm][0][aoff0];    // M-subtile 0 base
    const __bf16* aB1 = &sX[row1][0][aoff0];  // M-subtile 1 base (opaque row)

    v8f accQ0 = {}, accK0 = {}, accQ1 = {}, accK1 = {};
#pragma unroll 2
    for (int k0 = 0; k0 < D_; k0 += 32) {
      v16bf akh0 = ld_a(aB0);
      v16bf akl0 = ld_a(aB0 + LDSROW);
      v16bf aqh0 = ld_a(aB0 + 2 * LDSROW);
      v16bf aql0 = ld_a(aB0 + 3 * LDSROW);
      v16bf akh1 = ld_a(aB1);
      v16bf akl1 = ld_a(aB1 + LDSROW);
      v16bf aqh1 = ld_a(aB1 + 2 * LDSROW);
      v16bf aql1 = ld_a(aB1 + 3 * LDSROW);
      v16bf bqh = ld_b(pqh + k0);
      v16bf bql = ld_b(pql + k0);
      v16bf bkh = ld_b(pkh + k0);
      v16bf bkl = ld_b(pkl + k0);
      accQ0 = WMMA_BF16(aqh0, bqh, accQ0);
      accQ0 = WMMA_BF16(aqh0, bql, accQ0);
      accQ0 = WMMA_BF16(aql0, bqh, accQ0);
      accK0 = WMMA_BF16(akh0, bkh, accK0);
      accK0 = WMMA_BF16(akh0, bkl, accK0);
      accK0 = WMMA_BF16(akl0, bkh, accK0);
      accQ1 = WMMA_BF16(aqh1, bqh, accQ1);
      accQ1 = WMMA_BF16(aqh1, bql, accQ1);
      accQ1 = WMMA_BF16(aql1, bqh, accQ1);
      accK1 = WMMA_BF16(akh1, bkh, accK1);
      accK1 = WMMA_BF16(akh1, bkl, accK1);
      accK1 = WMMA_BF16(akl1, bkh, accK1);
      aB0 += 32;
      aB1 += 32;
    }
    // C/D layout: VGPR j, lanes 0-15 -> (m=j, n=lane), lanes 16-31 -> (m=j+8)
#pragma unroll
    for (int j = 0; j < 8; ++j) {
      float q0 = accQ0[j], k0v = accK0[j];
      psq[0][j]  += q0 * q0;
      psk[0][j]  += k0v * k0v;
      psqk[0][j] += q0 * k0v;
      float q1 = accQ1[j], k1v = accK1[j];
      psq[1][j]  += q1 * q1;
      psk[1][j]  += k1v * k1v;
      psqk[1][j] += q1 * k1v;
    }
  }
  // Reduce over the 16 lanes of each half-wave (n dimension).
#pragma unroll
  for (int s = 0; s < 2; ++s)
#pragma unroll
    for (int j = 0; j < 8; ++j) {
#pragma unroll
      for (int off = 1; off < 16; off <<= 1) {
        psq[s][j]  += __shfl_xor(psq[s][j], off, 32);
        psk[s][j]  += __shfl_xor(psk[s][j], off, 32);
        psqk[s][j] += __shfl_xor(psqk[s][j], off, 32);
      }
    }
  if (lm == 0) {  // lanes 0 (m=j) and 16 (m=j+8) carry the half-wave sums
#pragma unroll
    for (int s = 0; s < 2; ++s)
#pragma unroll
      for (int j = 0; j < 8; ++j) {
        int m = s * 16 + j + lhi * 8;
        atomicAdd(&sSq[m], psq[s][j]);
        atomicAdd(&sSk[m], psk[s][j]);
        atomicAdd(&sSqk[m], psqk[s][j]);
      }
  }
  __syncthreads();
  if (tid < MTILE) {
    int l = l0 + tid;
    float nq = fmaxf(sqrtf(sSq[tid]), 1e-12f);
    float nk = fmaxf(sqrtf(sSk[tid]), 1e-12f);
    float cs = sSqk[tid] / (nq * nk);
    float p  = fminf(fmaxf(0.5f * (1.f - cs), 0.f), 1.f);
    if (l == 0) p = 1.f;
    p_full[(size_t)b * L_ + l] = p;
  }
}

// ---------------------------------------------------------------------------
// Kernel 3 (one WG per batch): boundary scan, stable take_idx construction,
// chunk_id per token, and the 2048-step linear recurrence over chunks.
// ---------------------------------------------------------------------------
__global__ __launch_bounds__(1024) void k_chunk_scan(
    const float* __restrict__ x, const float* __restrict__ p_full,
    int* __restrict__ chunk_id, float* __restrict__ smoothed) {
  const int b   = blockIdx.x;
  const int tid = threadIdx.x;
  __shared__ int   sScan[1024];
  __shared__ int   sTake[C_];
  __shared__ float sCoef[C_];
  __shared__ float sDecay[C_];

  // Each thread owns 8 consecutive tokens.
  const int base = tid * 8;
  float pv[8];
  int bnd[8], csum[8];
  int s = 0;
#pragma unroll
  for (int i = 0; i < 8; ++i) {
    pv[i]   = p_full[(size_t)b * L_ + base + i];
    bnd[i]  = pv[i] >= 0.5f;
    s      += bnd[i];
    csum[i] = s;
  }
  sScan[tid] = s;
  __syncthreads();
  // Hillis-Steele inclusive scan of per-thread boundary counts.
  for (int off = 1; off < 1024; off <<= 1) {
    int v = (tid >= off) ? sScan[tid - off] : 0;
    __syncthreads();
    sScan[tid] += v;
    __syncthreads();
  }
  const int totalB = sScan[1023];
  const int excl   = (tid == 0) ? 0 : sScan[tid - 1];

#pragma unroll
  for (int i = 0; i < 8; ++i) {
    int l   = base + i;
    int cum = excl + csum[i];                 // inclusive #boundaries <= l
    chunk_id[(size_t)b * L_ + l] = min(cum - 1, C_ - 1);
    if (bnd[i]) {
      int r = cum - 1;                        // stable rank among boundaries
      if (r < C_) sTake[r] = l;
    } else {
      int slot = totalB + (l - cum);          // pad with non-boundary tokens
      if (slot < C_) sTake[slot] = l;
    }
  }
  __syncthreads();

  const int numC = min(totalB, C_);
  for (int c = tid; c < C_; c += 1024) {
    int idx  = sTake[c];
    float pc = fminf(fmaxf(p_full[(size_t)b * L_ + idx], 1e-4f), 1.f - 1e-4f);
    float coef = (c == 0) ? 1.f : pc;
    if (c >= numC) coef = 0.f;                // pad_mask zeroes chunked rows
    sCoef[c]  = coef;
    sDecay[c] = (c == 0) ? 0.f : (1.f - pc);
  }
  __syncthreads();

  // Linear recurrence: smoothed[c] = coef[c]*x[take[c]] + decay[c]*smoothed[c-1]
  // Thread = one d-lane; loads are independent, only the FMA chain is serial.
  float val = 0.f;
  const size_t xb = (size_t)b * L_ * D_;
  const size_t sb = (size_t)b * C_ * D_;
  for (int c = 0; c < C_; ++c) {
    float xv = x[xb + (size_t)sTake[c] * D_ + tid];
    val = sCoef[c] * xv + sDecay[c] * val;
    smoothed[sb + (size_t)c * D_ + tid] = val;
  }
}

// ---------------------------------------------------------------------------
// Kernel 4: out[b,l,:] = smoothed[b, chunk_id[b,l], :]   (streaming gather)
// ---------------------------------------------------------------------------
__global__ void k_gather_out(const float* __restrict__ smoothed,
                             const int* __restrict__ chunk_id,
                             float* __restrict__ out) {
  size_t i = (size_t)blockIdx.x * 256 + threadIdx.x;  // i < B*L*D
  int d       = (int)(i & (D_ - 1));
  size_t row  = i >> 10;            // b*L + l
  int b       = (int)(row >> 13);   // /8192
  int c       = chunk_id[row];
  out[i] = smoothed[(((size_t)b * C_) + c) * D_ + d];
}

// ---------------------------------------------------------------------------
extern "C" void kernel_launch(void* const* d_in, const int* in_sizes, int n_in,
                              void* d_out, int out_size, void* d_ws, size_t ws_size,
                              hipStream_t stream) {
  const float* x  = (const float*)d_in[0];
  const float* Wq = (const float*)d_in[1];
  const float* Wk = (const float*)d_in[2];
  // d_in[3] = chunk_divisor (== 4, baked into C_)

  char* ws = (char*)d_ws;
  size_t off = 0;
  auto take = [&](size_t bytes) {
    void* p = ws + off;
    off += (bytes + 255) & ~(size_t)255;
    return p;
  };
  __bf16* Wqh = (__bf16*)take((size_t)D_ * D_ * 2);
  __bf16* Wql = (__bf16*)take((size_t)D_ * D_ * 2);
  __bf16* Wkh = (__bf16*)take((size_t)D_ * D_ * 2);
  __bf16* Wkl = (__bf16*)take((size_t)D_ * D_ * 2);
  float*  p_full   = (float*)take((size_t)B_ * L_ * 4);
  int*    chunk_id = (int*)take((size_t)B_ * L_ * 4);
  float*  smoothed = (float*)take((size_t)B_ * C_ * D_ * 4);
  (void)ws_size; (void)n_in; (void)in_sizes; (void)out_size;

  k_wsplit<<<(D_ * D_ + 255) / 256, 256, 0, stream>>>(Wq, Wk, Wqh, Wql, Wkh, Wkl);
  k_gemm_p<<<B_ * (L_ / MTILE), 512, 0, stream>>>(x, Wqh, Wql, Wkh, Wkl, p_full);
  k_chunk_scan<<<B_, 1024, 0, stream>>>(x, p_full, chunk_id, smoothed);
  k_gather_out<<<(int)(((size_t)B_ * L_ * D_) / 256), 256, 0, stream>>>(
      smoothed, chunk_id, (float*)d_out);
}
